// VDNet_76536317215329
// MI455X (gfx1250) — matrix operand order repaired and checked by reading, not compiled, not run on hardware
//
#include <hip/hip_runtime.h>
#include <hip/hip_bf16.h>

typedef __attribute__((ext_vector_type(16))) _Float16 v16h;
typedef __attribute__((ext_vector_type(8)))  _Float16 v8h;
typedef __attribute__((ext_vector_type(8)))  float    v8f;

#define BN_EPS 1e-5f
#define TM 64
#define TN 128

// ---------------------------------------------------------------------------
// Weight prepack: f32 [Cout,K] -> f16 fragment-order chunks.
// Layout: wp[(row * nch + c)*32 + d], d = fragment slot within a 32-K chunk.
// A 16x32 f16 fragment: lane half h reads 16 contiguous halves at d = h*16+e,
// element e holds K = h*16+e (identity, matches B staging).
__global__ __launch_bounds__(256)
void pack_w_kernel(const float* __restrict__ w, _Float16* __restrict__ wp,
                   int Cout, int K, int CoutPad, int nch)
{
  int i = blockIdx.x * 256 + threadIdx.x;
  int total = CoutPad * nch * 32;
  if (i >= total) return;
  int d  = i & 31;
  int t  = i >> 5;
  int c  = t % nch;
  int row = t / nch;
  int dg = d >> 3;
  int g  = (dg == 1) ? 2 : (dg == 2) ? 1 : dg;   // self-inverse permute
  int k  = c * 32 + d;
  bool ok = (row < Cout) && (k < K);
  int idx = ok ? (row * K + k) : 0;
  float v = w[idx];
  wp[i] = (_Float16)(ok ? v : 0.f);
}

// ---------------------------------------------------------------------------
// Tiled implicit-GEMM conv (stride 1) via v_wmma_f32_16x16x32_f16.
// Block: 256 threads = 8 waves; block tile = 64(Cout) x 128(pixel).
// Per K-chunk each wave issues 4 WMMAs sharing one A fragment (direct 32B
// global load from prepacked f16 weights). B: im2col staged to LDS in
// fragment order (identity: K = half*16 + e); each thread stages 16
// contiguous K for one column with a division-free mixed-radix counter.
__global__ __launch_bounds__(256)
void conv2d_wmma_kernel(const float* __restrict__ x,
                        const _Float16* __restrict__ wp,
                        const float* __restrict__ bias,
                        float* __restrict__ y,
                        int Bn, int Cin, int Hin, int Win,
                        int Cout, int Hout, int Wout,
                        int kdim, int pad, int nch)
{
  __shared__ __align__(32) _Float16 Bs[TN * 32];

  const int tid  = threadIdx.x;
  const int lane = tid & 31;
  const int wave = tid >> 5;
  const int half = lane >> 4;
  const int l16  = lane & 15;

  const int K    = Cin * kdim * kdim;
  const int k2   = kdim * kdim;
  const int npix = Bn * Hout * Wout;
  const int tilesM = (Cout + TM - 1) / TM;
  const int bm  = (blockIdx.x % tilesM) * TM;
  const int bnn = (blockIdx.x / tilesM) * TN;

  // --- B staging assignment: 2 threads per column, 16 contiguous K each ---
  const int cb   = tid & 127;         // B column within tile
  const int gp   = tid >> 7;          // K half-chunk (0..1)
  const int ncol = bnn + cb;
  const bool colv = (ncol < npix);
  int nb, noy, nox;
  {
    int n = colv ? ncol : 0;
    nox = n % Wout;
    int t = n / Wout;
    noy = t % Hout;
    nb  = t / Hout;
  }

  // --- division-free im2col state for this thread's K walk ----------------
  int kcur = gp * 16;                 // first k of first chunk
  int cc = kcur / k2;
  int rr = kcur - cc * k2;
  int dy = rr / kdim;
  int dx = rr - dy * kdim;
  // per-chunk tail advance of +16 decomposed into counter digits
  const int q16  = 16 / k2;
  const int r16  = 16 - q16 * k2;
  const int dy16 = r16 / kdim;
  const int dx16 = r16 - dy16 * kdim;

  // --- wave compute assignment: wm in 0..3 (M), wn in 0..1 (N x4 subtiles)
  const int wm = wave & 3;
  const int wn = wave >> 2;
  const int arow = bm + wm * 16 + l16;            // always < CoutPad
  const _Float16* ap = wp + ((size_t)arow * nch) * 32 + half * 16;

  v8f acc[4] = {{}, {}, {}, {}};

  for (int ci = 0; ci < nch; ++ci) {
    // ---- stage B (im2col activations), branch/division-free ----------
    {
      v16h t;
#pragma unroll
      for (int e = 0; e < 16; ++e) {
        bool ok = colv && (kcur < K);
        int  iy = noy + dy - pad;
        int  ix = nox + dx - pad;
        ok = ok && iy >= 0 && iy < Hin && ix >= 0 && ix < Win;
        int idx0 = ((nb * Cin + cc) * Hin + iy) * Win + ix;
        int idx  = ok ? idx0 : 0;
        float v = x[idx];               // clamped index: pipelined load
        t[e] = (_Float16)(ok ? v : 0.f);
        // counter ++ : dx -> dy -> cc with conditional carries
        ++kcur;
        ++dx;
        bool c1 = (dx == kdim); dx = c1 ? 0 : dx; dy += c1 ? 1 : 0;
        bool c2 = (dy == kdim); dy = c2 ? 0 : dy; cc += c2 ? 1 : 0;
      }
      *(v16h*)&Bs[cb * 32 + gp * 16] = t;
      // advance counter by the remaining 16 of this 32-K chunk
      kcur += 16;
      dx += dx16;
      bool c1 = (dx >= kdim); dx = c1 ? dx - kdim : dx; dy += c1 ? 1 : 0;
      dy += dy16;
      bool c2 = (dy >= kdim); dy = c2 ? dy - kdim : dy; cc += c2 ? 1 : 0;
      cc += q16;
    }
    __syncthreads();

    // ---- compute: A direct from global (prepacked), four WMMAs -------
    __builtin_prefetch(ap + (ci + 1) * 32, 0, 1);
    v16h a = *(const v16h*)(ap + ci * 32);
#pragma unroll
    for (int j = 0; j < 4; ++j) {
      v16h b = *(const v16h*)&Bs[(wn * 64 + j * 16 + l16) * 32 + half * 16];
      acc[j] = __builtin_amdgcn_wmma_f32_16x16x32_f16(
          false, a, false, b, (short)0, acc[j], false, false);
    }
    __syncthreads();
  }

  // ---- store (C/D layout: M = half*8 + r, N = l16) -----------------------
#pragma unroll
  for (int j = 0; j < 4; ++j) {
    int n = bnn + wn * 64 + j * 16 + l16;
    if (n < npix) {
      int ox = n % Wout;
      int t  = n / Wout;
      int oy = t % Hout;
      int pb = t / Hout;
#pragma unroll
      for (int r = 0; r < 8; ++r) {
        int m = bm + wm * 16 + (half << 3) + r;
        if (m < Cout)
          y[((pb * Cout + m) * Hout + oy) * Wout + ox] = acc[j][r] + bias[m];
      }
    }
  }
}

// ---------------------------------------------------------------------------
// Training-mode batchnorm: per-channel mean/var over (N,H,W).
__global__ __launch_bounds__(256)
void bn_stats_kernel(const float* __restrict__ x, float* __restrict__ mv,
                     int C, int N, int HW)
{
  const int c = blockIdx.x;
  float s = 0.f, s2 = 0.f;
  for (int nn = 0; nn < N; ++nn) {
    const float* xp = x + (size_t)(nn * C + c) * HW;
    for (int p = threadIdx.x; p < HW; p += 256) {
      float v = xp[p];
      s += v; s2 += v * v;
    }
  }
  __shared__ float sh[256], sh2[256];
  sh[threadIdx.x] = s; sh2[threadIdx.x] = s2;
  __syncthreads();
  for (int st = 128; st > 0; st >>= 1) {
    if ((int)threadIdx.x < st) {
      sh[threadIdx.x]  += sh[threadIdx.x + st];
      sh2[threadIdx.x] += sh2[threadIdx.x + st];
    }
    __syncthreads();
  }
  if (threadIdx.x == 0) {
    float inv = 1.f / (float)(N * HW);
    float mu = sh[0] * inv;
    mv[c]     = mu;
    mv[C + c] = sh2[0] * inv - mu * mu;
  }
}

__global__ __launch_bounds__(256)
void bn_apply_kernel(float* __restrict__ x, const float* __restrict__ mv,
                     const float* __restrict__ g, const float* __restrict__ be,
                     int C, int N, int HW, int relu)
{
  int i = blockIdx.x * 256 + threadIdx.x;
  int total = N * C * HW;
  if (i >= total) return;
  int c = (i / HW) % C;
  float v = (x[i] - mv[c]) * rsqrtf(mv[C + c] + BN_EPS) * g[c] + be[c];
  if (relu) v = fmaxf(v, 0.f);
  x[i] = v;
}

// ---------------------------------------------------------------------------
// Max-pool s x s (s==1 acts as a copy).
__global__ __launch_bounds__(256)
void maxpool_kernel(const float* __restrict__ x, float* __restrict__ y,
                    int NC, int Hin, int Win, int s)
{
  int Hout = Hin / s, Wout = Win / s;
  int i = blockIdx.x * 256 + threadIdx.x;
  int total = NC * Hout * Wout;
  if (i >= total) return;
  int ox = i % Wout;
  int t  = i / Wout;
  int oy = t % Hout;
  int nc = t / Hout;
  float m = -3.402823466e38f;
  for (int dy = 0; dy < s; ++dy)
    for (int dx = 0; dx < s; ++dx)
      m = fmaxf(m, x[(nc * Hin + oy * s + dy) * Win + ox * s + dx]);
  y[i] = m;
}

// ---------------------------------------------------------------------------
// VAE: split h [B,6400] -> mean/logvar [B,3200], z = mean + exp(.5*lv)*eps.
__global__ __launch_bounds__(256)
void vae_kernel(const float* __restrict__ h, const float* __restrict__ eps,
                float* __restrict__ z, float* __restrict__ om,
                float* __restrict__ olv, int Bn)
{
  int i = blockIdx.x * 256 + threadIdx.x;
  int total = Bn * 3200;
  if (i >= total) return;
  int b = i / 3200, j = i - b * 3200;
  float mean = h[b * 6400 + j];
  float lv   = h[b * 6400 + 3200 + j];
  om[i]  = mean;
  olv[i] = lv;
  z[i] = mean + expf(0.5f * lv) * eps[i];
}

// ---------------------------------------------------------------------------
// Per-sample dynamic depthwise 5x5 conv (groups=32), kernels [B,4,32,1,5,5].
__global__ __launch_bounds__(256)
void dyn_dw_conv_kernel(const float* __restrict__ x, const float* __restrict__ ker,
                        float* __restrict__ y, int Bn, int H, int scaleIdx)
{
  int i = blockIdx.x * 256 + threadIdx.x;
  int total = Bn * 32 * H * H;
  if (i >= total) return;
  int ox = i % H;
  int t  = i / H;
  int oy = t % H;
  t /= H;
  int c = t % 32;
  int b = t / 32;
  const float* kk = ker + ((b * 4 + scaleIdx) * 32 + c) * 25;
  const float* xs = x + ((b * 32 + c) * H) * H;
  float s = 0.f;
  for (int dy = 0; dy < 5; ++dy) {
    int iy = oy + dy - 2;
    if (iy < 0 || iy >= H) continue;
    for (int dx = 0; dx < 5; ++dx) {
      int ix = ox + dx - 2;
      if (ix < 0 || ix >= H) continue;
      s += xs[iy * H + ix] * kk[dy * 5 + dx];
    }
  }
  y[i] = s;
}

// ---------------------------------------------------------------------------
// Nearest-upsample each scale to 64x64 and concat to [B,128,64,64].
__global__ __launch_bounds__(256)
void upcat_kernel(const float* __restrict__ f0, const float* __restrict__ f1,
                  const float* __restrict__ f2, const float* __restrict__ f3,
                  float* __restrict__ y, int Bn)
{
  int i = blockIdx.x * 256 + threadIdx.x;
  int total = Bn * 128 * 64 * 64;
  if (i >= total) return;
  int ox = i & 63;
  int t  = i >> 6;
  int oy = t & 63;
  t >>= 6;
  int c = t % 128;
  int b = t / 128;
  int k  = c >> 5;
  int cl = c & 31;
  const float* f = (k == 0) ? f0 : (k == 1) ? f1 : (k == 2) ? f2 : f3;
  int S  = 8 << k;
  int sf = 64 / S;
  y[i] = f[((b * 32 + cl) * S + oy / sf) * S + ox / sf];
}

// ---------------------------------------------------------------------------
extern "C" void kernel_launch(void* const* d_in, const int* in_sizes, int n_in,
                              void* d_out, int out_size, void* d_ws, size_t ws_size,
                              hipStream_t stream) {
  const int Bn = in_sizes[0] / (3 * 32 * 32);

  // ---- workspace bump allocator -------------------------------------------
  char* ws = (char*)d_ws;
  size_t off = 0;
  auto alloc = [&](size_t nbytes) -> void* {
    void* p = (void*)(ws + off);
    off = (off + nbytes + 255) & ~(size_t)255;
    return p;
  };
  auto allocf = [&](size_t n) -> float* { return (float*)alloc(n * 4); };

  const int Sfeat[4] = {8, 16, 32, 64};
  float* feats[4]; float* outs[4];
  for (int k = 0; k < 4; ++k) feats[k] = allocf((size_t)Bn * 32 * Sfeat[k] * Sfeat[k]);
  for (int k = 0; k < 4; ++k) outs[k]  = allocf((size_t)Bn * 32 * Sfeat[k] * Sfeat[k]);
  float* stats = allocf(512);
  float* henc  = allocf((size_t)Bn * 256 * 64);    // [B,256,8,8]
  float* hfin  = allocf((size_t)Bn * 256 * 25);    // [B,256,5,5]
  float* z     = allocf((size_t)Bn * 3200);
  float* kz1   = allocf((size_t)Bn * 3200);
  float* kz2   = allocf((size_t)Bn * 3200);
  float* xcat  = allocf((size_t)Bn * 128 * 64 * 64);
  _Float16* wpack = (_Float16*)alloc((size_t)4 * 1024 * 1024 * 2); // 4M halves
  const size_t bigN = (size_t)Bn * 64 * 256 * 256; // largest conv output
  float* buf0 = allocf(bigN);
  float* buf1 = allocf(bigN);
  (void)ws_size; (void)n_in;

  float* out_x    = (float*)d_out;                 // [B,3,64,64]
  float* out_mean = out_x + (size_t)Bn * 3 * 64 * 64;
  float* out_lv   = out_mean + (size_t)Bn * 3200;

  // ---- launch helpers ------------------------------------------------------
  auto conv = [&](const float* x, const float* w, const float* b, float* y,
                  int Cin, int Hin, int Cout, int Hout, int kd, int pad) {
    int K    = Cin * kd * kd;
    int nch  = (K + 31) / 32;
    int tilesM  = (Cout + TM - 1) / TM;
    int CoutPad = tilesM * TM;
    int ptotal  = CoutPad * nch * 32;
    pack_w_kernel<<<(ptotal + 255) / 256, 256, 0, stream>>>(
        w, wpack, Cout, K, CoutPad, nch);
    int npix  = Bn * Hout * Hout;
    int tiles = tilesM * ((npix + TN - 1) / TN);
    conv2d_wmma_kernel<<<tiles, 256, 0, stream>>>(
        x, wpack, b, y, Bn, Cin, Hin, Hin, Cout, Hout, Hout, kd, pad, nch);
  };
  auto bn = [&](float* x, const float* g, const float* be, int C, int HW, int relu) {
    bn_stats_kernel<<<C, 256, 0, stream>>>(x, stats, C, Bn, HW);
    int total = Bn * C * HW;
    bn_apply_kernel<<<(total + 255) / 256, 256, 0, stream>>>(x, stats, g, be, C, Bn, HW, relu);
  };
  auto pool = [&](const float* x, float* y, int C, int Hin, int s) {
    int Hout = Hin / s;
    int total = Bn * C * Hout * Hout;
    maxpool_kernel<<<(total + 255) / 256, 256, 0, stream>>>(x, y, Bn * C, Hin, Hin, s);
  };

  int p = 6;  // param tensors start after i0..i3, m, eps
  auto nextp = [&]() -> const float* { return (const float*)d_in[p++]; };

  // ---- image encoders ------------------------------------------------------
  const int ich[5]   = {3, 64, 64, 64, 32};
  const int isamp[4] = {2, 1, 2, 1};
  const int S0s[4]   = {32, 64, 128, 256};
  for (int k = 0; k < 4; ++k) {
    const float* src = (const float*)d_in[k];
    int H = S0s[k];
    for (int l = 0; l < 4; ++l) {
      const float* w  = nextp();
      const float* b  = nextp();
      const float* g  = nextp();
      const float* be = nextp();
      float* cbuf = (src == buf0) ? buf1 : buf0;
      float* obuf = (cbuf == buf0) ? buf1 : buf0;
      conv(src, w, b, cbuf, ich[l], H, ich[l + 1], H, 5, 2);
      bn(cbuf, g, be, ich[l + 1], H * H, 1);
      int s = isamp[l];
      float* dst = (l == 3) ? feats[k] : obuf;
      pool(cbuf, dst, ich[l + 1], H, s);
      H /= s;
      src = dst;
    }
  }

  // ---- motion encoder ------------------------------------------------------
  const int mch[7]   = {6, 96, 96, 128, 128, 256, 256};
  const int msamp[6] = {4, 1, 2, 1, 2, 1};
  {
    const float* src = (const float*)d_in[4];
    int H = 128;
    for (int l = 0; l < 6; ++l) {
      const float* w  = nextp();
      const float* b  = nextp();
      const float* g  = nextp();
      const float* be = nextp();
      float* cbuf = (src == buf0) ? buf1 : buf0;
      float* obuf = (cbuf == buf0) ? buf1 : buf0;
      conv(src, w, b, cbuf, mch[l], H, mch[l + 1], H, 5, 2);
      bn(cbuf, g, be, mch[l + 1], H * H, 1);
      int s = msamp[l];
      float* dst = (l == 5) ? henc : obuf;
      pool(cbuf, dst, mch[l + 1], H, s);
      H /= s;
      src = dst;
    }
  }
  // motion_final: 4x4 conv, pad 0 : [B,256,8,8] -> [B,256,5,5]
  {
    const float* w = nextp();
    const float* b = nextp();
    conv(henc, w, b, hfin, 256, 8, 256, 5, 4, 0);
  }

  // ---- VAE sample ----------------------------------------------------------
  {
    int total = Bn * 3200;
    vae_kernel<<<(total + 255) / 256, 256, 0, stream>>>(
        hfin, (const float*)d_in[5], z, out_mean, out_lv, Bn);
  }

  // ---- kernel decoder ------------------------------------------------------
  {
    const float* w0 = nextp(); const float* b0 = nextp();
    const float* g0 = nextp(); const float* be0 = nextp();
    conv(z, w0, b0, kz1, 128, 5, 128, 5, 5, 2);
    bn(kz1, g0, be0, 128, 25, 1);
    const float* w1 = nextp(); const float* b1 = nextp();
    nextp(); nextp();  // layer-1 g/beta unused (last layer, no bn/relu)
    conv(kz1, w1, b1, kz2, 128, 5, 128, 5, 5, 2);
    const float* gk = nextp(); const float* bek = nextp();  // kernel_bn
    bn(kz2, gk, bek, 128, 25, 0);
  }

  // ---- per-sample dynamic depthwise convs ---------------------------------
  for (int k = 0; k < 4; ++k) {
    int total = Bn * 32 * Sfeat[k] * Sfeat[k];
    dyn_dw_conv_kernel<<<(total + 255) / 256, 256, 0, stream>>>(
        feats[k], kz2, outs[k], Bn, Sfeat[k], k);
  }

  // ---- upsample + concat ---------------------------------------------------
  {
    int total = Bn * 128 * 64 * 64;
    upcat_kernel<<<(total + 255) / 256, 256, 0, stream>>>(
        outs[0], outs[1], outs[2], outs[3], xcat, Bn);
  }

  // ---- motion decoder ------------------------------------------------------
  {
    const float* w0 = nextp(); const float* b0 = nextp();
    const float* g0 = nextp(); const float* be0 = nextp();
    conv(xcat, w0, b0, buf0, 128, 64, 128, 64, 9, 4);
    bn(buf0, g0, be0, 128, 64 * 64, 1);
    const float* w1 = nextp(); const float* b1 = nextp();
    const float* g1 = nextp(); const float* be1 = nextp();
    conv(buf0, w1, b1, buf1, 128, 64, 128, 64, 1, 0);
    bn(buf1, g1, be1, 128, 64 * 64, 1);
    const float* w2 = nextp(); const float* b2 = nextp();
    nextp(); nextp();  // last layer: no bn/relu
    conv(buf1, w2, b2, out_x, 128, 64, 3, 64, 1, 0);
  }
}
